// QuantumTransformerLayer_39960375722237
// MI455X (gfx1250) — matrix operand order repaired and checked by reading, not compiled, not run on hardware
//
#include <hip/hip_runtime.h>
#include <hip/hip_bf16.h>
#include <math.h>

typedef __attribute__((ext_vector_type(16))) _Float16 v16h;
typedef __attribute__((ext_vector_type(8)))  _Float16 v8h;
typedef __attribute__((ext_vector_type(8)))  float    v8f;

// Problem constants (from reference): B=2, S=2048, D=1024, H=16, dk=64, F=4096
#define BATCH 2
#define SEQ   2048
#define DMODEL 1024
#define NHEAD 16
#define DHEAD 64
#define FFDIM 4096
#define MROWS 4096   // B*S

// ---------------------------------------------------------------------------
// WMMA fragment loaders (CDNA5 16x16x32 f16 layouts, CDNA5 ISA 7.12.2)
// ---------------------------------------------------------------------------
// A fragment: row m = lane%16, half = lane/16.
//   elem j (j<8):  k = k0 + half*8 + j
//   elem j (j>=8): k = k0 + 16 + half*8 + (j-8)
__device__ inline v16h frag_a_f16(const _Float16* __restrict__ Abase, int lda, int k0) {
  const int lane = threadIdx.x & 31;
  const int m = lane & 15, hf = lane >> 4;
  const _Float16* p = Abase + (size_t)m * lda + k0 + hf * 8;
  v8h lo = *(const v8h*)p;
  v8h hi = *(const v8h*)(p + 16);
  v16h r;
#pragma unroll
  for (int i = 0; i < 8; ++i) { r[i] = lo[i]; r[i + 8] = hi[i]; }
  return r;
}

// Same A layout but source is fp32 (attention probabilities); convert in regs.
__device__ inline v16h frag_a_f32(const float* __restrict__ Abase, int lda, int k0) {
  const int lane = threadIdx.x & 31;
  const int m = lane & 15, hf = lane >> 4;
  const float* p = Abase + (size_t)m * lda + k0 + hf * 8;
  v8f lo = *(const v8f*)p;
  v8f hi = *(const v8f*)(p + 16);
  v16h r;
#pragma unroll
  for (int i = 0; i < 8; ++i) { r[i] = (_Float16)lo[i]; r[i + 8] = (_Float16)hi[i]; }
  return r;
}

// B fragment: B = W^T, so column n of B is row n of W (row-major, ld = K).
//   col n = lane%16, half = lane/16, elem j<8 -> k = k0 + half*8 + j, j>=8 -> k0 + 16 + half*8 + (j-8)
__device__ inline v16h frag_b_f16(const _Float16* __restrict__ Wbase, int ldw, int k0) {
  const int lane = threadIdx.x & 31;
  const int n = lane & 15, hf = lane >> 4;
  const _Float16* p = Wbase + (size_t)n * ldw + k0 + hf * 8;
  v8h lo = *(const v8h*)p;
  v8h hi = *(const v8h*)(p + 16);
  v16h r;
#pragma unroll
  for (int i = 0; i < 8; ++i) { r[i] = lo[i]; r[i + 8] = hi[i]; }
  return r;
}

__device__ inline v8f wmma_step(v16h a, v16h b, v8f c) {
  return __builtin_amdgcn_wmma_f32_16x16x32_f16(false, a, false, b, (short)0, c,
                                                false, false);
}

#define V8F_ZERO {0.f, 0.f, 0.f, 0.f, 0.f, 0.f, 0.f, 0.f}

// ---------------------------------------------------------------------------
// Generic GEMM: C[M,N] = A16[M,K] @ W16[N,K]^T + bias, with fused epilogues.
// LD = leading dimension of both A and W (compile-time: immediate offsets).
// Block = 256 threads = 8 waves; wave w owns a 16x64 strip (4 accumulators).
// K-loop unrolled by 2 with ping-pong fragment sets: each set is written only
// by its loads, so there is no register rotation (no v_mov chains), and each
// 4-WMMA group overlaps the other set's in-flight loads. K % 64 == 0 for all
// call sites. Loop conditions are scalar -> EXEC stays all-ones.
//  mode 0: store f16              (Q/K/V projections)
//  mode 1: store f32              (attention output projection)
//  mode 2: quantum-FF + ReLU, f16 (first FFN GEMM)
//  mode 3: * cos(layer_phase), f32 (second FFN GEMM)
// ---------------------------------------------------------------------------
template <int LD>
__global__ __launch_bounds__(256) void gemm_wmma(
    const _Float16* __restrict__ A, const _Float16* __restrict__ W,
    const float* __restrict__ bias, int K, int mode,
    _Float16* __restrict__ outH, float* __restrict__ outF, int ldo,
    const float* __restrict__ ffPh, const float* __restrict__ ffWr,
    const float* __restrict__ ffWi, const float* __restrict__ layerPhase) {
  const int wave = threadIdx.x >> 5, lane = threadIdx.x & 31;
  const int m0 = blockIdx.y * 128 + wave * 16;
  const int n0 = blockIdx.x * 64;
  const _Float16* Ab = A + (size_t)m0 * LD;
  const _Float16* Wb = W + (size_t)n0 * LD;
  const size_t wstep = (size_t)16 * LD;

  v8f c0 = V8F_ZERO, c1 = V8F_ZERO, c2 = V8F_ZERO, c3 = V8F_ZERO;
  // Ping set (even slabs)
  v16h a  = frag_a_f16(Ab, LD, 0);
  v16h b0 = frag_b_f16(Wb, LD, 0);
  v16h b1 = frag_b_f16(Wb + wstep, LD, 0);
  v16h b2 = frag_b_f16(Wb + 2 * wstep, LD, 0);
  v16h b3 = frag_b_f16(Wb + 3 * wstep, LD, 0);

  for (int k0 = 0;;) {
    // Pong set loads (k0+32 always valid: K % 64 == 0, k0 <= K-64)
    v16h a1  = frag_a_f16(Ab, LD, k0 + 32);
    v16h b10 = frag_b_f16(Wb, LD, k0 + 32);
    v16h b11 = frag_b_f16(Wb + wstep, LD, k0 + 32);
    v16h b12 = frag_b_f16(Wb + 2 * wstep, LD, k0 + 32);
    v16h b13 = frag_b_f16(Wb + 3 * wstep, LD, k0 + 32);

    c0 = wmma_step(a, b0, c0);       // 4 independent v_wmma on ping set
    c1 = wmma_step(a, b1, c1);
    c2 = wmma_step(a, b2, c2);
    c3 = wmma_step(a, b3, c3);

    const bool more = (k0 + 64) < K; // scalar
    if (more) {                      // refill ping set (k0+64)
      __builtin_prefetch(Ab + (size_t)(lane & 15) * LD + k0 + 96, 0, 1);
      a  = frag_a_f16(Ab, LD, k0 + 64);
      b0 = frag_b_f16(Wb, LD, k0 + 64);
      b1 = frag_b_f16(Wb + wstep, LD, k0 + 64);
      b2 = frag_b_f16(Wb + 2 * wstep, LD, k0 + 64);
      b3 = frag_b_f16(Wb + 3 * wstep, LD, k0 + 64);
    }
    c0 = wmma_step(a1, b10, c0);     // 4 independent v_wmma on pong set
    c1 = wmma_step(a1, b11, c1);
    c2 = wmma_step(a1, b12, c2);
    c3 = wmma_step(a1, b13, c3);
    if (!more) break;
    k0 += 64;
  }

  const int nl = lane & 15, hf = lane >> 4;
  const float scale = (mode == 3) ? __cosf(layerPhase[0]) : 1.f;
  auto emit = [&](v8f c, int nt) {
    const int n = n0 + nt * 16 + nl;
    const float bn = bias[n];
    float phase_ff = 0.f, wi_n = 0.f, ph_n = 0.f;
    if (mode == 2) {
      ph_n = ffPh[n]; wi_n = ffWi[n];
      float sp, cp;
      __sincosf(ph_n, &sp, &cp);
      phase_ff = cp * ffWr[n] + sp * wi_n;
    }
#pragma unroll
    for (int r = 0; r < 8; ++r) {
      const int m = m0 + r + 8 * hf;
      float v = c[r] + bn;
      if (mode == 2) {
        const float imag_ff = wi_n * __sinf(ph_n + v * 0.1f);   // v_sin_f32
        v = v + phase_ff * imag_ff * 0.1f;
        v = v > 0.f ? v : 0.f;
      }
      if (mode == 3) v *= scale;
      if (mode == 0 || mode == 2) outH[(size_t)m * ldo + n] = (_Float16)v;
      else                        outF[(size_t)m * ldo + n] = v;
    }
  };
  emit(c0, 0); emit(c1, 1); emit(c2, 2); emit(c3, 3);
}

// ---------------------------------------------------------------------------
// Attention scores: enhanced = s*cos(theta + 0.1*s) + 0.1*imag, s = QK^T/8.
// Wave owns 16x64 of the (2048x2048) per-(b,h) score matrix; Q fragments
// (K=64 -> two slabs, fully resident) reused across 4 key tiles = 8 v_wmma.
// cos via hardware v_cos_f32 (args are O(10): safely inside accurate range) —
// this epilogue runs on 134M elements, libm's Payne-Hanek path would dominate.
// Writes fp32 directly into the attn_w region of d_out (softmax'd in place).
// ---------------------------------------------------------------------------
__global__ __launch_bounds__(256) void attn_scores(
    const _Float16* __restrict__ Q16, const _Float16* __restrict__ K16,
    const float* __restrict__ phases, const float* __restrict__ imagH,
    float* __restrict__ attnW) {
  const int bh = blockIdx.z, b = bh >> 4, h = bh & 15;
  const int wave = threadIdx.x >> 5, lane = threadIdx.x & 31;
  const int m0 = blockIdx.y * 128 + wave * 16;
  const int n0 = blockIdx.x * 64;
  const _Float16* Ab = Q16 + (size_t)b * SEQ * DMODEL + h * DHEAD + (size_t)m0 * DMODEL;
  const _Float16* Wb = K16 + (size_t)b * SEQ * DMODEL + h * DHEAD + (size_t)n0 * DMODEL;

  const v16h a0 = frag_a_f16(Ab, DMODEL, 0);
  const v16h a1 = frag_a_f16(Ab, DMODEL, 32);

  const float theta = phases[h];
  const float im = imagH[h] * 0.1f;
  float* orow = attnW + (size_t)bh * SEQ * SEQ;
  const int nl = lane & 15, hf = lane >> 4;

#pragma unroll
  for (int nt = 0; nt < 4; ++nt) {
    const _Float16* Wt = Wb + (size_t)(nt * 16) * DMODEL;
    v16h bk0 = frag_b_f16(Wt, DMODEL, 0);
    v16h bk1 = frag_b_f16(Wt, DMODEL, 32);
    v8f c = V8F_ZERO;
    c = wmma_step(a0, bk0, c);
    c = wmma_step(a1, bk1, c);
    const int n = n0 + nt * 16 + nl;
#pragma unroll
    for (int r = 0; r < 8; ++r) {
      const int m = m0 + r + 8 * hf;
      const float s = c[r] * 0.125f;  // 1/sqrt(64)
      orow[(size_t)m * SEQ + n] = s * __cosf(theta + s * 0.1f) + im;
    }
  }
}

// Row softmax over 2048, in place (one block per row, 8 elems/thread).
// exp via hardware v_exp_f32 (arguments are <= 0 after max-subtraction).
__global__ __launch_bounds__(256) void softmax_rows(float* __restrict__ attn) {
  float* p = attn + (size_t)blockIdx.x * SEQ;
  __shared__ float red[256];
  const int t = threadIdx.x;
  float v[8], mx = -3.402823e38f;
#pragma unroll
  for (int i = 0; i < 8; ++i) { v[i] = p[t + i * 256]; mx = fmaxf(mx, v[i]); }
  red[t] = mx; __syncthreads();
  for (int s = 128; s > 0; s >>= 1) { if (t < s) red[t] = fmaxf(red[t], red[t + s]); __syncthreads(); }
  mx = red[0]; __syncthreads();
  float sum = 0.f;
#pragma unroll
  for (int i = 0; i < 8; ++i) { v[i] = __expf(v[i] - mx); sum += v[i]; }
  red[t] = sum; __syncthreads();
  for (int s = 128; s > 0; s >>= 1) { if (t < s) red[t] += red[t + s]; __syncthreads(); }
  const float inv = 1.f / red[0];
#pragma unroll
  for (int i = 0; i < 8; ++i) p[t + i * 256] = v[i] * inv;
}

// attended[b,s,h*64+d] = P(b,h) @ V_h. Wave owns 16 rows x all 64 head dims
// (4 accumulators). A = fp32 probs converted in-register while assembling the
// fragment; B columns contiguous via V^T. Ping-pong pipelined 2048-deep K loop.
__global__ __launch_bounds__(256) void attended_gemm(
    const float* __restrict__ P, const _Float16* __restrict__ Vt16,
    _Float16* __restrict__ att16) {
  const int bh = blockIdx.z, b = bh >> 4, h = bh & 15;
  const int wave = threadIdx.x >> 5, lane = threadIdx.x & 31;
  const int m0 = blockIdx.y * 128 + wave * 16;
  const float* Pb = P + (size_t)bh * SEQ * SEQ + (size_t)m0 * SEQ;
  const _Float16* Wb = Vt16 + (size_t)bh * DHEAD * SEQ;
  const size_t wstep = (size_t)16 * SEQ;

  v8f c0 = V8F_ZERO, c1 = V8F_ZERO, c2 = V8F_ZERO, c3 = V8F_ZERO;
  v16h a  = frag_a_f32(Pb, SEQ, 0);
  v16h b0 = frag_b_f16(Wb, SEQ, 0);
  v16h b1 = frag_b_f16(Wb + wstep, SEQ, 0);
  v16h b2 = frag_b_f16(Wb + 2 * wstep, SEQ, 0);
  v16h b3 = frag_b_f16(Wb + 3 * wstep, SEQ, 0);

  for (int k0 = 0;;) {
    v16h a1  = frag_a_f32(Pb, SEQ, k0 + 32);
    v16h b10 = frag_b_f16(Wb, SEQ, k0 + 32);
    v16h b11 = frag_b_f16(Wb + wstep, SEQ, k0 + 32);
    v16h b12 = frag_b_f16(Wb + 2 * wstep, SEQ, k0 + 32);
    v16h b13 = frag_b_f16(Wb + 3 * wstep, SEQ, k0 + 32);

    c0 = wmma_step(a, b0, c0);
    c1 = wmma_step(a, b1, c1);
    c2 = wmma_step(a, b2, c2);
    c3 = wmma_step(a, b3, c3);

    const bool more = (k0 + 64) < SEQ;
    if (more) {
      __builtin_prefetch(Pb + (size_t)(lane & 15) * SEQ + k0 + 96, 0, 1);
      a  = frag_a_f32(Pb, SEQ, k0 + 64);
      b0 = frag_b_f16(Wb, SEQ, k0 + 64);
      b1 = frag_b_f16(Wb + wstep, SEQ, k0 + 64);
      b2 = frag_b_f16(Wb + 2 * wstep, SEQ, k0 + 64);
      b3 = frag_b_f16(Wb + 3 * wstep, SEQ, k0 + 64);
    }
    c0 = wmma_step(a1, b10, c0);
    c1 = wmma_step(a1, b11, c1);
    c2 = wmma_step(a1, b12, c2);
    c3 = wmma_step(a1, b13, c3);
    if (!more) break;
    k0 += 64;
  }

  const int nl = lane & 15, hf = lane >> 4;
  auto emit = [&](v8f c, int nt) {
#pragma unroll
    for (int r = 0; r < 8; ++r) {
      const int m = m0 + r + 8 * hf;
      att16[((size_t)b * SEQ + m) * DMODEL + h * DHEAD + nt * 16 + nl] = (_Float16)c[r];
    }
  };
  emit(c0, 0); emit(c1, 1); emit(c2, 2); emit(c3, 3);
}

// LayerNorm(resid + add) * g + be ; writes fp32 (+ optional f16 copy).
__global__ __launch_bounds__(256) void ln_kernel(
    const float* __restrict__ resid, const float* __restrict__ add,
    const float* __restrict__ g, const float* __restrict__ be,
    float* __restrict__ outF, _Float16* __restrict__ outH) {
  const size_t row = blockIdx.x;
  const float* r0 = resid + row * DMODEL;
  const float* a0 = add + row * DMODEL;
  __shared__ float red[256];
  const int t = threadIdx.x;
  float x[4], s = 0.f;
#pragma unroll
  for (int i = 0; i < 4; ++i) { x[i] = r0[t + i * 256] + a0[t + i * 256]; s += x[i]; }
  red[t] = s; __syncthreads();
  for (int st = 128; st > 0; st >>= 1) { if (t < st) red[t] += red[t + st]; __syncthreads(); }
  const float mu = red[0] * (1.f / DMODEL); __syncthreads();
  float vs = 0.f;
#pragma unroll
  for (int i = 0; i < 4; ++i) { const float d = x[i] - mu; vs += d * d; }
  red[t] = vs; __syncthreads();
  for (int st = 128; st > 0; st >>= 1) { if (t < st) red[t] += red[t + st]; __syncthreads(); }
  const float inv = rsqrtf(red[0] * (1.f / DMODEL) + 1e-5f);
#pragma unroll
  for (int i = 0; i < 4; ++i) {
    const int col = t + i * 256;
    const float y = (x[i] - mu) * inv * g[col] + be[col];
    outF[row * DMODEL + col] = y;
    if (outH) outH[row * DMODEL + col] = (_Float16)y;
  }
}

__global__ void cast_f32_f16(const float* __restrict__ in, _Float16* __restrict__ out, int n) {
  const int i = blockIdx.x * blockDim.x + threadIdx.x;
  if (i < n) out[i] = (_Float16)in[i];
}

// Vt[b,h,d,s] = V[b,s,h*64+d]  (makes B-fragment columns contiguous)
__global__ void transpose_v(const _Float16* __restrict__ V16, _Float16* __restrict__ Vt16) {
  const size_t tid = (size_t)blockIdx.x * blockDim.x + threadIdx.x;
  const int d = (int)(tid & (DMODEL - 1));
  const size_t sg = tid >> 10;          // global row 0..4095
  const int b = (int)(sg >> 11);        // /SEQ
  const int sloc = (int)(sg & (SEQ - 1));
  const int h = d >> 6, dd = d & 63;
  Vt16[(((size_t)(b * NHEAD + h) * DHEAD + dd) << 11) + sloc] = V16[tid];
}

// imag[h] = sum_d(wr[h,d]*wi[h,d]) * sin(phase[h])
__global__ void prep_imag(const float* __restrict__ wr, const float* __restrict__ wi,
                          const float* __restrict__ ph, float* __restrict__ imag) {
  __shared__ float red[64];
  const int h = blockIdx.x, t = threadIdx.x;
  red[t] = wr[h * DHEAD + t] * wi[h * DHEAD + t]; __syncthreads();
  for (int s = 32; s > 0; s >>= 1) { if (t < s) red[t] += red[t + s]; __syncthreads(); }
  if (t == 0) imag[h] = red[0] * __sinf(ph[h]);
}

// ---------------------------------------------------------------------------
extern "C" void kernel_launch(void* const* d_in, const int* in_sizes, int n_in,
                              void* d_out, int out_size, void* d_ws, size_t ws_size,
                              hipStream_t stream) {
  const float* src   = (const float*)d_in[0];
  const float* Wq    = (const float*)d_in[1];
  const float* bq    = (const float*)d_in[2];
  const float* Wk    = (const float*)d_in[3];
  const float* bk    = (const float*)d_in[4];
  const float* Wv    = (const float*)d_in[5];
  const float* bv    = (const float*)d_in[6];
  const float* Wo    = (const float*)d_in[7];
  const float* bo    = (const float*)d_in[8];
  const float* aPh   = (const float*)d_in[9];
  const float* aWr   = (const float*)d_in[10];
  const float* aWi   = (const float*)d_in[11];
  const float* W1    = (const float*)d_in[12];
  const float* b1    = (const float*)d_in[13];
  const float* W2    = (const float*)d_in[14];
  const float* b2    = (const float*)d_in[15];
  const float* ffPh  = (const float*)d_in[16];
  const float* ffWr  = (const float*)d_in[17];
  const float* ffWi  = (const float*)d_in[18];
  const float* g1    = (const float*)d_in[19];
  const float* be1   = (const float*)d_in[20];
  const float* g2    = (const float*)d_in[21];
  const float* be2   = (const float*)d_in[22];
  const float* layerPhase = (const float*)d_in[23];

  float* outMain = (float*)d_out;                           // (B,S,D)
  float* attnW   = (float*)d_out + (size_t)MROWS * DMODEL;  // (B,H,S,S), softmax'd in place

  // Workspace carve-out (all sizes 256B-aligned by construction)
  char* ws = (char*)d_ws;
  size_t off = 0;
  auto carve = [&](size_t bytes) { void* p = ws + off; off += (bytes + 255) & ~(size_t)255; return p; };
  _Float16* src16 = (_Float16*)carve((size_t)MROWS * DMODEL * 2);
  _Float16* Wq16  = (_Float16*)carve((size_t)DMODEL * DMODEL * 2);
  _Float16* Wk16  = (_Float16*)carve((size_t)DMODEL * DMODEL * 2);
  _Float16* Wv16  = (_Float16*)carve((size_t)DMODEL * DMODEL * 2);
  _Float16* Wo16  = (_Float16*)carve((size_t)DMODEL * DMODEL * 2);
  _Float16* W116  = (_Float16*)carve((size_t)FFDIM * DMODEL * 2);
  _Float16* W216  = (_Float16*)carve((size_t)DMODEL * FFDIM * 2);
  _Float16* Q16   = (_Float16*)carve((size_t)MROWS * DMODEL * 2);
  _Float16* K16   = (_Float16*)carve((size_t)MROWS * DMODEL * 2);
  _Float16* V16   = (_Float16*)carve((size_t)MROWS * DMODEL * 2);
  _Float16* Vt16  = (_Float16*)carve((size_t)MROWS * DMODEL * 2);
  _Float16* att16 = (_Float16*)carve((size_t)MROWS * DMODEL * 2);
  float*    attnO = (float*)carve((size_t)MROWS * DMODEL * 4);
  float*    xF    = (float*)carve((size_t)MROWS * DMODEL * 4);
  _Float16* x16   = (_Float16*)carve((size_t)MROWS * DMODEL * 2);
  _Float16* qh16  = (_Float16*)carve((size_t)MROWS * FFDIM * 2);
  float*    ffF   = (float*)carve((size_t)MROWS * DMODEL * 4);
  float*    imagH = (float*)carve(256);

  // 1) fp32 -> f16 casts (activations + all weights)
  auto cast = [&](const float* in, _Float16* out16, int n) {
    cast_f32_f16<<<(n + 255) / 256, 256, 0, stream>>>(in, out16, n);
  };
  cast(src, src16, MROWS * DMODEL);
  cast(Wq, Wq16, DMODEL * DMODEL);
  cast(Wk, Wk16, DMODEL * DMODEL);
  cast(Wv, Wv16, DMODEL * DMODEL);
  cast(Wo, Wo16, DMODEL * DMODEL);
  cast(W1, W116, FFDIM * DMODEL);
  cast(W2, W216, DMODEL * FFDIM);
  prep_imag<<<NHEAD, 64, 0, stream>>>(aWr, aWi, aPh, imagH);

  const dim3 blk(256);
  // 2) Q/K/V projections (mode 0: f16 out). Wave strip = 16x64.
  const dim3 gProj(DMODEL / 64, MROWS / 128);
  gemm_wmma<DMODEL><<<gProj, blk, 0, stream>>>(src16, Wq16, bq, DMODEL, 0,
      Q16, nullptr, DMODEL, nullptr, nullptr, nullptr, nullptr);
  gemm_wmma<DMODEL><<<gProj, blk, 0, stream>>>(src16, Wk16, bk, DMODEL, 0,
      K16, nullptr, DMODEL, nullptr, nullptr, nullptr, nullptr);
  gemm_wmma<DMODEL><<<gProj, blk, 0, stream>>>(src16, Wv16, bv, DMODEL, 0,
      V16, nullptr, DMODEL, nullptr, nullptr, nullptr, nullptr);
  transpose_v<<<(MROWS * DMODEL) / 256, 256, 0, stream>>>(V16, Vt16);

  // 3) scores + quantum enhancement fused (writes attn region of d_out)
  attn_scores<<<dim3(SEQ / 64, SEQ / 128, BATCH * NHEAD), blk, 0, stream>>>(
      Q16, K16, aPh, imagH, attnW);
  // 4) softmax in place
  softmax_rows<<<BATCH * NHEAD * SEQ, blk, 0, stream>>>(attnW);
  // 5) attended = P @ V (fp32 probs converted in-register)
  attended_gemm<<<dim3(1, SEQ / 128, BATCH * NHEAD), blk, 0, stream>>>(
      attnW, Vt16, att16);
  // 6) output projection (mode 1: f32 out)
  gemm_wmma<DMODEL><<<gProj, blk, 0, stream>>>(att16, Wo16, bo, DMODEL, 1,
      nullptr, attnO, DMODEL, nullptr, nullptr, nullptr, nullptr);
  // 7) x = LN(src + attn_out)  (f32 + f16 copies)
  ln_kernel<<<MROWS, blk, 0, stream>>>(src, attnO, g1, be1, xF, x16);
  // 8) qh = relu(quantum(x @ W1^T + b1))  (mode 2: fused epilogue, f16 out)
  gemm_wmma<DMODEL><<<dim3(FFDIM / 64, MROWS / 128), blk, 0, stream>>>(
      x16, W116, b1, DMODEL, 2,
      qh16, nullptr, FFDIM, ffPh, ffWr, ffWi, nullptr);
  // 9) ff = (qh @ W2^T + b2) * cos(layer_phase)  (mode 3: f32 out)
  gemm_wmma<FFDIM><<<gProj, blk, 0, stream>>>(qh16, W216, b2, FFDIM, 3,
      nullptr, ffF, DMODEL, nullptr, nullptr, nullptr, layerPhase);
  // 10) out = LN(x + ff)
  ln_kernel<<<MROWS, blk, 0, stream>>>(xF, ffF, g2, be2, outMain, nullptr);
}
